// DiscreteCommunication_88837103551524
// MI455X (gfx1250) — matrix operand order repaired and checked by reading, not compiled, not run on hardware
//
#include <hip/hip_runtime.h>

// ---------------------------------------------------------------------------
// DiscreteCommunication forward for MI455X (gfx1250, wave32, WMMA bf16)
// ---------------------------------------------------------------------------

typedef __attribute__((ext_vector_type(8)))  __bf16 v8bf;
typedef __attribute__((ext_vector_type(16))) __bf16 v16bf;
typedef __attribute__((ext_vector_type(8)))  float  v8f;

#define N_NODES 20000
#define HID     128
#define TWO_MSG 128
#define KIN     256           // 2*HID (encoder input / GRU x input)
#define EPS_G   1e-10f

#define MASK_WORDS (N_NODES * 4)          // 4 x u32 bitmask per node
#define FEAT_ELEMS (N_NODES * HID)
#define WENC_ELEMS (TWO_MSG * KIN)        // 128x256
#define WIH_ELEMS  (3 * HID * KIN)        // 384x256
#define WHH_ELEMS  (3 * HID * HID)        // 384x128
#define WDEC_ELEMS (TWO_MSG * TWO_MSG)    // 128x128

// workspace byte offsets (all multiples of 256)
#define OFF_MASK  ((size_t)0)
#define OFF_FEAT  ((size_t)320000)
#define OFF_H     (OFF_FEAT + (size_t)FEAT_ELEMS * 2)
#define OFF_WENC  (OFF_H    + (size_t)FEAT_ELEMS * 2)
#define OFF_WIH   (OFF_WENC + (size_t)WENC_ELEMS * 2)
#define OFF_WHH   (OFF_WIH  + (size_t)WIH_ELEMS  * 2)
#define OFF_WDEC  (OFF_WHH  + (size_t)WHH_ELEMS  * 2)
#define OFF_CDEC  (OFF_WDEC + (size_t)WDEC_ELEMS * 2)

#define WMMA_BF16(a, b, c) \
  __builtin_amdgcn_wmma_f32_16x16x32_bf16(false, (a), false, (b), (short)0, (c), false, false)

// A fragment (16x32 bf16): lane<16 -> K {k0..k0+7, k0+16..k0+23} with k0=kk*32,
// lane>=16 -> same +8.  p points at row + kk*32 + (lane>>4)*8.
static __device__ __forceinline__ v16bf load_a_frag(const __bf16* p) {
  v16bf a;
  *((v8bf*)&a)       = *((const v8bf*)p);
  *(((v8bf*)&a) + 1) = *((const v8bf*)(p + 16));
  return a;
}
// B fragment (32x16 bf16): lane<16 -> col=lane, K {k0..k0+7, k0+16..k0+23} with k0=kk*32;
// lane>=16 -> col=lane-16, same +8.  p = weightRow(col) + kk*32 + (lane>>4)*8.
static __device__ __forceinline__ v16bf load_b_frag(const __bf16* p) {
  v16bf b;
  *((v8bf*)&b)       = *((const v8bf*)p);
  *(((v8bf*)&b) + 1) = *((const v8bf*)(p + 16));
  return b;
}

static __device__ __forceinline__ float sigmoidf_(float x) {
  return 1.0f / (1.0f + __expf(-x));
}

// ---------------------------------------------------------------------------
// Kernel 0: zero node bitmasks, convert feat/h/weights fp32 -> bf16
// ---------------------------------------------------------------------------
__global__ void prep_kernel(const float* __restrict__ feat, const float* __restrict__ h,
                            const float* __restrict__ W_enc, const float* __restrict__ W_ih,
                            const float* __restrict__ W_hh,  const float* __restrict__ W_dec,
                            unsigned* __restrict__ maskBuf,
                            __bf16* __restrict__ featbf, __bf16* __restrict__ hbf,
                            __bf16* __restrict__ Wencbf, __bf16* __restrict__ Wihbf,
                            __bf16* __restrict__ Whhbf,  __bf16* __restrict__ Wdecbf) {
  long t = (long)blockIdx.x * blockDim.x + threadIdx.x;
  if (t < MASK_WORDS) { maskBuf[t] = 0u; return; }
  t -= MASK_WORDS;
  if (t < FEAT_ELEMS) { featbf[t] = (__bf16)feat[t]; return; }
  t -= FEAT_ELEMS;
  if (t < FEAT_ELEMS) { hbf[t] = (__bf16)h[t]; return; }
  t -= FEAT_ELEMS;
  if (t < WENC_ELEMS) { Wencbf[t] = (__bf16)W_enc[t]; return; }
  t -= WENC_ELEMS;
  if (t < WIH_ELEMS)  { Wihbf[t] = (__bf16)W_ih[t]; return; }
  t -= WIH_ELEMS;
  if (t < WHH_ELEMS)  { Whhbf[t] = (__bf16)W_hh[t]; return; }
  t -= WHH_ELEMS;
  if (t < WDEC_ELEMS) { Wdecbf[t] = (__bf16)W_dec[t]; return; }
}
#define PREP_TOTAL (MASK_WORDS + 2 * FEAT_ELEMS + WENC_ELEMS + WIH_ELEMS + WHH_ELEMS + WDEC_ELEMS)

// ---------------------------------------------------------------------------
// Kernel 1: per-edge logits (WMMA), gumbel hard-argmax, atomicOr into node mask
// block = 128 threads (4 waves), 16 edges; wave w owns output cols [32w,32w+32)
// = bitmask word w.
// ---------------------------------------------------------------------------
__global__ __launch_bounds__(128) void edge_kernel(const __bf16* __restrict__ featbf,
                                                   const __bf16* __restrict__ hbf,
                                                   const __bf16* __restrict__ Wencbf,
                                                   const float* __restrict__ b_enc,
                                                   const float* __restrict__ u,
                                                   const int* __restrict__ src,
                                                   const int* __restrict__ dst,
                                                   unsigned* __restrict__ maskBuf) {
  const int lane = threadIdx.x & 31;
  const int wv   = threadIdx.x >> 5;   // 0..3
  const int lm   = lane & 15;
  const int hi16 = lane >> 4;          // 0|1
  const long eBase = (long)blockIdx.x * 16;

  const int arow = src[eBase + lm];
  const __bf16* aF = featbf + (long)arow * HID;
  const __bf16* aH = hbf    + (long)arow * HID;

  const int col0 = wv * 32 + lm;
  const int col1 = col0 + 16;
  const __bf16* bR0 = Wencbf + (long)col0 * KIN;   // B^T row n == W_enc row n
  const __bf16* bR1 = Wencbf + (long)col1 * KIN;

  v8f acc0 = {}; v8f acc1 = {};
#pragma unroll
  for (int kk = 0; kk < 8; ++kk) {
    const int kA = kk * 32 + hi16 * 8;
    const int kB = kk * 32 + hi16 * 8;
    const __bf16* ap = (kA < HID) ? (aF + kA) : (aH + (kA - HID));
    v16bf a  = load_a_frag(ap);
    v16bf b0 = load_b_frag(bR0 + kB);
    v16bf b1 = load_b_frag(bR1 + kB);
    acc0 = WMMA_BF16(a, b0, acc0);
    acc1 = WMMA_BF16(a, b1, acc1);
  }

  const float bias0 = b_enc[col0];
  const float bias1 = b_enc[col1];
  const bool odd = (lm & 1);

#pragma unroll
  for (int r = 0; r < 8; ++r) {
    const long e = eBase + r + hi16 * 8;          // edge handled by this lane half
    const float u0 = u[e * TWO_MSG + col0];
    const float u1 = u[e * TWO_MSG + col1];
    const float g0 = -__logf(-__logf(u0 + EPS_G) + EPS_G);
    const float g1 = -__logf(-__logf(u1 + EPS_G) + EPS_G);
    const float y0 = acc0[r] + bias0 + g0;
    const float y1 = acc1[r] + bias1 + g1;
    const float o0 = __shfl_xor(y0, 1, 32);       // partner column in the pair
    const float o1 = __shfl_xor(y1, 1, 32);
    // argmax semantics: index 0 wins ties
    const bool p0 = odd ? (y0 > o0) : (y0 >= o0);
    const bool p1 = odd ? (y1 > o1) : (y1 >= o1);
    const unsigned m0 = __builtin_amdgcn_ballot_w32(p0);
    const unsigned m1 = __builtin_amdgcn_ballot_w32(p1);
    // lanes 0-15 = edge r, lanes 16-31 = edge r+8; word w = cols [32w,32w+32)
    const unsigned wordLo = (m0 & 0xFFFFu) | ((m1 & 0xFFFFu) << 16);
    const unsigned wordHi = (m0 >> 16)     | (m1 & 0xFFFF0000u);
    if (lane == 0)  atomicOr(maskBuf + (long)dst[eBase + r]     * 4 + wv, wordLo);
    if (lane == 16) atomicOr(maskBuf + (long)dst[eBase + r + 8] * 4 + wv, wordHi);
  }
}

// ---------------------------------------------------------------------------
// Kernel 2: cdec = c @ W_dec^T + b_dec  (A generated from node bitmasks)
// block = 128 threads, 16 nodes; result stored bf16.
// ---------------------------------------------------------------------------
__global__ __launch_bounds__(128) void cdec_kernel(const unsigned* __restrict__ maskBuf,
                                                   const __bf16* __restrict__ Wdecbf,
                                                   const float* __restrict__ b_dec,
                                                   __bf16* __restrict__ cdecbf) {
  const int lane = threadIdx.x & 31;
  const int wv   = threadIdx.x >> 5;
  const int lm   = lane & 15;
  const int hi16 = lane >> 4;
  const int nodeBase = blockIdx.x * 16;

  const unsigned* mrow = maskBuf + (long)(nodeBase + lm) * 4;
  const int off = hi16 * 8;

  v16bf afrag[4];
#pragma unroll
  for (int kk = 0; kk < 4; ++kk) {
    const unsigned w = mrow[kk];
#pragma unroll
    for (int j = 0; j < 8; ++j) {
      afrag[kk][j]     = (__bf16)(((w >> (off + j))      & 1u) ? 1.0f : 0.0f);
      afrag[kk][8 + j] = (__bf16)(((w >> (off + 16 + j)) & 1u) ? 1.0f : 0.0f);
    }
  }

#pragma unroll
  for (int ct = 0; ct < 2; ++ct) {
    const int col = wv * 32 + ct * 16 + lm;
    const __bf16* bRow = Wdecbf + (long)col * TWO_MSG;
    v8f acc = {};
#pragma unroll
    for (int kk = 0; kk < 4; ++kk)
      acc = WMMA_BF16(afrag[kk], load_b_frag(bRow + kk * 32 + hi16 * 8), acc);
    const float bias = b_dec[col];
#pragma unroll
    for (int r = 0; r < 8; ++r) {
      const long m = nodeBase + r + hi16 * 8;
      cdecbf[m * HID + col] = (__bf16)(acc[r] + bias);
    }
  }
}

// ---------------------------------------------------------------------------
// Kernel 3: fused GRU.  gi = [feat,cdec] @ W_ih^T, gh = h @ W_hh^T, then
// elementwise GRU; writes h_new to both output halves.
// block = 128 threads, 16 nodes; wave w owns cols [32w,32w+32).
// ---------------------------------------------------------------------------
__global__ __launch_bounds__(128) void node_kernel(const __bf16* __restrict__ featbf,
                                                   const __bf16* __restrict__ cdecbf,
                                                   const __bf16* __restrict__ hbf,
                                                   const __bf16* __restrict__ Wihbf,
                                                   const __bf16* __restrict__ Whhbf,
                                                   const float* __restrict__ b_ih,
                                                   const float* __restrict__ b_hh,
                                                   const float* __restrict__ h_in,
                                                   float* __restrict__ out, long NH) {
  const int lane = threadIdx.x & 31;
  const int wv   = threadIdx.x >> 5;
  const int lm   = lane & 15;
  const int hi16 = lane >> 4;
  const int nodeBase = blockIdx.x * 16;

  const long node = nodeBase + lm;
  const __bf16* aF = featbf + node * HID;
  const __bf16* aC = cdecbf + node * HID;
  const __bf16* aH = hbf    + node * HID;

#pragma unroll
  for (int ct = 0; ct < 2; ++ct) {
    const int col = wv * 32 + ct * 16 + lm;
    const __bf16* bIR = Wihbf + (long)col * KIN;
    const __bf16* bIZ = Wihbf + (long)(col + HID) * KIN;
    const __bf16* bIN = Wihbf + (long)(col + 2 * HID) * KIN;
    const __bf16* bHR = Whhbf + (long)col * HID;
    const __bf16* bHZ = Whhbf + (long)(col + HID) * HID;
    const __bf16* bHN = Whhbf + (long)(col + 2 * HID) * HID;

    v8f giR = {}, giZ = {}, giN = {}, ghR = {}, ghZ = {}, ghN = {};
#pragma unroll
    for (int kk = 0; kk < 8; ++kk) {              // x = [feat, cdec], K = 256
      const int kA = kk * 32 + hi16 * 8;
      const int kB = kk * 32 + hi16 * 8;
      const __bf16* ap = (kA < HID) ? (aF + kA) : (aC + (kA - HID));
      v16bf a = load_a_frag(ap);
      giR = WMMA_BF16(a, load_b_frag(bIR + kB), giR);
      giZ = WMMA_BF16(a, load_b_frag(bIZ + kB), giZ);
      giN = WMMA_BF16(a, load_b_frag(bIN + kB), giN);
    }
#pragma unroll
    for (int kk = 0; kk < 4; ++kk) {              // h, K = 128
      const int kA = kk * 32 + hi16 * 8;
      const int kB = kk * 32 + hi16 * 8;
      v16bf a = load_a_frag(aH + kA);
      ghR = WMMA_BF16(a, load_b_frag(bHR + kB), ghR);
      ghZ = WMMA_BF16(a, load_b_frag(bHZ + kB), ghZ);
      ghN = WMMA_BF16(a, load_b_frag(bHN + kB), ghN);
    }

    const float biR = b_ih[col],           bhR = b_hh[col];
    const float biZ = b_ih[col + HID],     bhZ = b_hh[col + HID];
    const float biN = b_ih[col + 2 * HID], bhN = b_hh[col + 2 * HID];

#pragma unroll
    for (int r = 0; r < 8; ++r) {
      const long m = nodeBase + r + hi16 * 8;
      const float rg = sigmoidf_(giR[r] + ghR[r] + biR + bhR);
      const float zg = sigmoidf_(giZ[r] + ghZ[r] + biZ + bhZ);
      const float nt = tanhf((giN[r] + biN) + rg * (ghN[r] + bhN));
      const float ho = h_in[m * HID + col];
      const float hn = (1.0f - zg) * nt + zg * ho;
      out[m * HID + col]      = hn;
      out[NH + m * HID + col] = hn;   // tuple (h_new, h_new)
    }
  }
}

// ---------------------------------------------------------------------------
extern "C" void kernel_launch(void* const* d_in, const int* in_sizes, int n_in,
                              void* d_out, int out_size, void* d_ws, size_t ws_size,
                              hipStream_t stream) {
  const float* feat  = (const float*)d_in[0];
  const float* h     = (const float*)d_in[1];
  const int*   src   = (const int*)d_in[2];
  const int*   dst   = (const int*)d_in[3];
  const float* u     = (const float*)d_in[4];
  const float* W_enc = (const float*)d_in[5];
  const float* b_enc = (const float*)d_in[6];
  const float* W_dec = (const float*)d_in[7];
  const float* b_dec = (const float*)d_in[8];
  const float* W_ih  = (const float*)d_in[9];
  const float* W_hh  = (const float*)d_in[10];
  const float* b_ih  = (const float*)d_in[11];
  const float* b_hh  = (const float*)d_in[12];

  const int E = in_sizes[2];                 // 320000
  const int N = in_sizes[0] / HID;           // 20000
  const long NH = (long)N * HID;

  char* ws = (char*)d_ws;
  unsigned* maskBuf = (unsigned*)(ws + OFF_MASK);
  __bf16*   featbf  = (__bf16*)(ws + OFF_FEAT);
  __bf16*   hbf     = (__bf16*)(ws + OFF_H);
  __bf16*   Wencbf  = (__bf16*)(ws + OFF_WENC);
  __bf16*   Wihbf   = (__bf16*)(ws + OFF_WIH);
  __bf16*   Whhbf   = (__bf16*)(ws + OFF_WHH);
  __bf16*   Wdecbf  = (__bf16*)(ws + OFF_WDEC);
  __bf16*   cdecbf  = (__bf16*)(ws + OFF_CDEC);

  {
    const int threads = 256;
    const int blocks = (PREP_TOTAL + threads - 1) / threads;
    prep_kernel<<<blocks, threads, 0, stream>>>(feat, h, W_enc, W_ih, W_hh, W_dec,
                                                maskBuf, featbf, hbf, Wencbf, Wihbf,
                                                Whhbf, Wdecbf);
  }
  edge_kernel<<<E / 16, 128, 0, stream>>>(featbf, hbf, Wencbf, b_enc, u, src, dst, maskBuf);
  cdec_kernel<<<N / 16, 128, 0, stream>>>(maskBuf, Wdecbf, b_dec, cdecbf);
  node_kernel<<<N / 16, 128, 0, stream>>>(featbf, cdecbf, hbf, Wihbf, Whhbf,
                                          b_ih, b_hh, h, (float*)d_out, NH);
}
